// CausalSelfAttention_23089744183345
// MI455X (gfx1250) — matrix operand order repaired and checked by reading, not compiled, not run on hardware
//
#include <hip/hip_runtime.h>

typedef __bf16 bf16;
typedef __attribute__((ext_vector_type(16))) __bf16 v16bf;
typedef __attribute__((ext_vector_type(8)))  __bf16 bf16x8;
typedef __attribute__((ext_vector_type(8)))  float  v8f;

#define B_   2
#define T_   2048
#define C_   2048
#define NH_  16
#define NKV_ 4
#define HD_  128
#define M_   (B_ * T_)
#define NKVD (NKV_ * HD_)   // 512

static __device__ __forceinline__ v16bf join8(bf16x8 lo, bf16x8 hi) {
  return __builtin_shufflevector(lo, hi, 0,1,2,3,4,5,6,7,8,9,10,11,12,13,14,15);
}
static __device__ __forceinline__ v8f wmma_bf16(v16bf a, v16bf b, v8f c) {
  return __builtin_amdgcn_wmma_f32_16x16x32_bf16(false, a, false, b, (short)0, c, false, false);
}
// async copy: global -> LDS, 16B per lane, tracked by ASYNCcnt
static __device__ __forceinline__ void async_b128(const void* gaddr, void* lds) {
  unsigned loff = (unsigned)(unsigned long long)lds;  // low 32 bits = LDS offset
  asm volatile("global_load_async_to_lds_b128 %0, %1, off"
               :: "v"(loff), "v"(gaddr) : "memory");
}
static __device__ __forceinline__ void async_wait0() {
  asm volatile("s_wait_asynccnt 0x0" ::: "memory");
}

// ---------------------------------------------------------------- converts
__global__ void f32_to_bf16_kernel(const float* __restrict__ in,
                                   bf16* __restrict__ out, int n) {
  int i = blockIdx.x * blockDim.x + threadIdx.x;
  if (i < n) out[i] = (bf16)in[i];
}

// Wt[n][k] = (bf16) W[k][n]   (W is K x N row-major)
__global__ void transpose_to_bf16_kernel(const float* __restrict__ W,
                                         bf16* __restrict__ Wt, int K, int N) {
  int i = blockIdx.x * blockDim.x + threadIdx.x;
  if (i >= K * N) return;
  int n = i / K;
  int k = i - n * K;
  Wt[i] = (bf16)W[(long)k * N + n];
}

// ---------------------------------------------------------------- GEMM
// out[M][N] = A[M][K] * Bt[N][K]^T ; A,Bt bf16 ; out f32 or bf16
// block tile 128x256, 8 waves as 2(m) x 4(n), each wave 64x64 (16 accum frags)
#define BM 128
#define BN 256
#define BK 64
#define APAD 72   // 64+8 bf16 -> 144B row stride: 16B aligned, conflict-free b128

__global__ __launch_bounds__(256)
void gemm_bf16_kernel(const bf16* __restrict__ A, const bf16* __restrict__ Bt,
                      void* __restrict__ outv, int M, int N, int K, int out_f32) {
  __shared__ bf16 As[2][BM][APAD];
  __shared__ bf16 Bs[2][BN][APAD];

  const int tid   = threadIdx.x;
  const int lane  = tid & 31;
  const int wid   = tid >> 5;      // 0..7
  const int wm    = wid >> 2;      // 0..1 -> 64 rows
  const int wn    = wid & 3;       // 0..3 -> 64 cols
  const int khalf = lane >> 4;
  const int l16   = lane & 15;

  const int m0 = blockIdx.y * BM;
  const int n0 = blockIdx.x * BN;

  // staging: 64-wide rows, 8 chunks of 8 bf16; 256 threads
  const int chunk = tid & 7;
  const int rbase = tid >> 3;  // 0..31

  auto issue_tile = [&](int k0, int buf) {
#pragma unroll
    for (int rr = 0; rr < 4; ++rr) {           // A: 128 rows
      int r = rbase + rr * 32;
      async_b128(A + (long)(m0 + r) * K + k0 + chunk * 8, &As[buf][r][chunk * 8]);
    }
#pragma unroll
    for (int rr = 0; rr < 8; ++rr) {           // B: 256 rows
      int r = rbase + rr * 32;
      async_b128(Bt + (long)(n0 + r) * K + k0 + chunk * 8, &Bs[buf][r][chunk * 8]);
    }
  };

  v8f acc[4][4];
#pragma unroll
  for (int i = 0; i < 4; ++i)
#pragma unroll
    for (int j = 0; j < 4; ++j) acc[i][j] = (v8f)0.0f;

  const int kTiles = K / BK;
  issue_tile(0, 0);

  for (int kt = 0; kt < kTiles; ++kt) {
    const int cur = kt & 1;
    async_wait0();
    __syncthreads();
    if (kt + 1 < kTiles) {
      issue_tile((kt + 1) * BK, cur ^ 1);
      if (kt + 2 < kTiles) {  // L2 prefetch two tiles ahead
        __builtin_prefetch(A  + (long)(m0 + rbase) * K + (kt + 2) * BK + chunk * 8, 0, 1);
        __builtin_prefetch(Bt + (long)(n0 + rbase) * K + (kt + 2) * BK + chunk * 8, 0, 1);
      }
    }
#pragma unroll
    for (int ks = 0; ks < 2; ++ks) {
      v16bf afrag[4], bfrag[4];
#pragma unroll
      for (int i = 0; i < 4; ++i) {
        int r = wm * 64 + i * 16 + l16;
        afrag[i] = join8(*(const bf16x8*)&As[cur][r][ks * 32 + khalf * 8],
                         *(const bf16x8*)&As[cur][r][ks * 32 + 16 + khalf * 8]);
      }
#pragma unroll
      for (int j = 0; j < 4; ++j) {
        int c = wn * 64 + j * 16 + l16;
        bfrag[j] = join8(*(const bf16x8*)&Bs[cur][c][ks * 32 + khalf * 8],
                         *(const bf16x8*)&Bs[cur][c][ks * 32 + 16 + khalf * 8]);
      }
#pragma unroll
      for (int i = 0; i < 4; ++i)
#pragma unroll
        for (int j = 0; j < 4; ++j)
          acc[i][j] = wmma_bf16(afrag[i], bfrag[j], acc[i][j]);
    }
  }

#pragma unroll
  for (int i = 0; i < 4; ++i)
#pragma unroll
    for (int j = 0; j < 4; ++j)
#pragma unroll
      for (int r = 0; r < 8; ++r) {
        int m = m0 + wm * 64 + i * 16 + r + 8 * khalf;
        int n = n0 + wn * 64 + j * 16 + l16;
        float v = acc[i][j][r];
        if (out_f32) ((float*)outv)[(long)m * N + n] = v;
        else         ((bf16*)outv)[(long)m * N + n] = (bf16)v;
      }
}

// ---------------------------------------------------------------- RoPE
// buf: [M][nheads*128] bf16, head dim 128 split as (x1 | x2) halves of 64
__global__ void rope_kernel(bf16* __restrict__ buf, int nheads, float scale) {
  int i = blockIdx.x * blockDim.x + threadIdx.x;
  int per = nheads * 64;
  if (i >= M_ * per) return;
  int rowi = i / per;
  int rem  = i - rowi * per;
  int head = rem >> 6;
  int j    = rem & 63;
  int t    = rowi & (T_ - 1);
  // inv_freq = 10000^(-j/64) = exp(-j * ln(10000)/64)
  float inv_freq = __expf(-(float)j * 0.14391156544511f);
  float ang = (float)t * inv_freq;
  float c, s;
  __sincosf(ang, &s, &c);
  bf16* p = buf + (long)rowi * (nheads * HD_) + head * HD_ + j;
  float x1 = (float)p[0];
  float x2 = (float)p[64];
  p[0]  = (bf16)((x1 * c - x2 * s) * scale);
  p[64] = (bf16)((x2 * c + x1 * s) * scale);
}

// ---------------------------------------------------------------- flash attn
// 256 threads = 8 waves, each wave owns 16 q-rows; KV tiles of 64 shared.
#define BQ   128
#define BKV  64
#define KPAD 136   // 128+8 -> 272B rows
#define VPAD 72    // 64+8  -> 144B rows
#define PPAD 72

__global__ __launch_bounds__(256)
void flash_attn_kernel(const bf16* __restrict__ Q, const bf16* __restrict__ Kb,
                       const bf16* __restrict__ Vb, bf16* __restrict__ Y) {
  __shared__ bf16 Qs[BQ][KPAD];
  __shared__ bf16 Ks[2][BKV][KPAD];
  __shared__ bf16 Vts[2][HD_][VPAD];     // transposed: [d][kv]
  __shared__ bf16 Ps[8][16][PPAD];       // per-wave P staging

  const int tid   = threadIdx.x;   // 0..255 (8 waves)
  const int lane  = tid & 31;
  const int w     = tid >> 5;      // 0..7
  const int khalf = lane >> 4;
  const int l16   = lane & 15;

  const int q0  = blockIdx.x * BQ;
  const int h   = blockIdx.y;
  const int b   = blockIdx.z;
  const int kvh = h / (NH_ / NKV_);

  const bf16* Qbase = Q  + ((long)(b * T_ + q0)) * C_ + h * HD_;
  const bf16* Kbase = Kb + ((long)(b * T_)) * NKVD + kvh * HD_;
  const bf16* Vbase = Vb + ((long)(b * T_)) * NKVD + kvh * HD_;

  // KV staging: 4 iters x (256 threads x 8 bf16) covers 64x128
  auto issue_K = [&](int kv0, int buf) {
#pragma unroll
    for (int it = 0; it < 4; ++it) {
      int idx = tid * 8 + it * 2048;
      int r = idx >> 7, c = idx & 127;
      async_b128(Kbase + (long)(kv0 + r) * NKVD + c, &Ks[buf][r][c]);
    }
  };
  bf16x8 vreg[4];
  auto load_V = [&](int kv0) {
#pragma unroll
    for (int it = 0; it < 4; ++it) {
      int idx = tid * 8 + it * 2048;
      int r = idx >> 7, c = idx & 127;
      vreg[it] = *(const bf16x8*)(Vbase + (long)(kv0 + r) * NKVD + c);
    }
  };
  auto store_Vt = [&](int buf) {
#pragma unroll
    for (int it = 0; it < 4; ++it) {
      int idx = tid * 8 + it * 2048;
      int r = idx >> 7, c = idx & 127;
#pragma unroll
      for (int e = 0; e < 8; ++e) Vts[buf][c + e][r] = vreg[it][e];
    }
  };

  // prologue: stage Q (async, 128x128), K tile 0 (async), V tile 0 (regs -> LDS)
#pragma unroll
  for (int it = 0; it < 8; ++it) {
    int idx = tid * 8 + it * 2048;
    int r = idx >> 7, c = idx & 127;
    async_b128(Qbase + (long)r * C_ + c, &Qs[r][c]);
  }
  issue_K(0, 0);
  load_V(0);
  store_Vt(0);
  async_wait0();
  __syncthreads();

  v16bf qfrag[4];
#pragma unroll
  for (int d = 0; d < 4; ++d) {
    int r = w * 16 + l16;
    qfrag[d] = join8(*(const bf16x8*)&Qs[r][d * 32 + khalf * 8],
                     *(const bf16x8*)&Qs[r][d * 32 + 16 + khalf * 8]);
  }

  v8f o[8];
#pragma unroll
  for (int t = 0; t < 8; ++t) o[t] = (v8f)0.0f;
  float m_i[8], l_i[8];
#pragma unroll
  for (int r = 0; r < 8; ++r) { m_i[r] = -__builtin_inff(); l_i[r] = 0.0f; }

  const int ntiles = q0 / BKV + 2;   // diagonal spans two KV tiles for BQ=128
  for (int jt = 0; jt < ntiles; ++jt) {
    const int cur = jt & 1;
    const int kv0 = jt * BKV;
    if (jt + 1 < ntiles) {          // overlap next tile's loads with compute
      issue_K((jt + 1) * BKV, cur ^ 1);
      load_V((jt + 1) * BKV);
    }

    // S = Q K^T  (per wave: 16 q-rows x 64 kv)
    v8f s[4];
#pragma unroll
    for (int j = 0; j < 4; ++j) s[j] = (v8f)0.0f;
#pragma unroll
    for (int d = 0; d < 4; ++d)
#pragma unroll
      for (int j = 0; j < 4; ++j) {
        int c = j * 16 + l16;
        v16bf kf = join8(*(const bf16x8*)&Ks[cur][c][d * 32 + khalf * 8],
                         *(const bf16x8*)&Ks[cur][c][d * 32 + 16 + khalf * 8]);
        s[j] = wmma_bf16(qfrag[d], kf, s[j]);
      }

    if (kv0 + BKV - 1 > q0) {  // tile may cross the causal diagonal
#pragma unroll
      for (int j = 0; j < 4; ++j)
#pragma unroll
        for (int r = 0; r < 8; ++r) {
          int qrow = q0 + w * 16 + r + 8 * khalf;
          int kcol = kv0 + j * 16 + l16;
          if (kcol > qrow) s[j][r] = -3.0e38f;
        }
    }

    float mnew[8], alpha[8];
#pragma unroll
    for (int r = 0; r < 8; ++r) {
      float v = fmaxf(fmaxf(s[0][r], s[1][r]), fmaxf(s[2][r], s[3][r]));
#pragma unroll
      for (int off = 1; off < 16; off <<= 1) v = fmaxf(v, __shfl_xor(v, off, 32));
      mnew[r]  = fmaxf(m_i[r], v);
      alpha[r] = __expf(m_i[r] - mnew[r]);
      m_i[r]   = mnew[r];
    }
#pragma unroll
    for (int j = 0; j < 4; ++j)
#pragma unroll
      for (int r = 0; r < 8; ++r) s[j][r] = __expf(s[j][r] - mnew[r]);
#pragma unroll
    for (int r = 0; r < 8; ++r) {
      float v = s[0][r] + s[1][r] + s[2][r] + s[3][r];
#pragma unroll
      for (int off = 1; off < 16; off <<= 1) v += __shfl_xor(v, off, 32);
      l_i[r] = l_i[r] * alpha[r] + v;
    }
#pragma unroll
    for (int t = 0; t < 8; ++t)
#pragma unroll
      for (int r = 0; r < 8; ++r) o[t][r] *= alpha[r];

    // D-layout P -> LDS -> A-layout fragments (same-wave, DS-counter ordered)
#pragma unroll
    for (int j = 0; j < 4; ++j)
#pragma unroll
      for (int r = 0; r < 8; ++r)
        Ps[w][r + 8 * khalf][j * 16 + l16] = (bf16)s[j][r];

    v16bf pf[2];
#pragma unroll
    for (int c = 0; c < 2; ++c)
      pf[c] = join8(*(const bf16x8*)&Ps[w][l16][c * 32 + khalf * 8],
                    *(const bf16x8*)&Ps[w][l16][c * 32 + 16 + khalf * 8]);

#pragma unroll
    for (int t = 0; t < 8; ++t)
#pragma unroll
      for (int c = 0; c < 2; ++c) {
        int dc = t * 16 + l16;
        v16bf vf = join8(*(const bf16x8*)&Vts[cur][dc][c * 32 + khalf * 8],
                         *(const bf16x8*)&Vts[cur][dc][c * 32 + 16 + khalf * 8]);
        o[t] = wmma_bf16(pf[c], vf, o[t]);
      }

    if (jt + 1 < ntiles) store_Vt(cur ^ 1);   // after compute: alt buffer is free
    async_wait0();
    __syncthreads();
  }

  float invl[8];
#pragma unroll
  for (int r = 0; r < 8; ++r) invl[r] = 1.0f / l_i[r];
#pragma unroll
  for (int t = 0; t < 8; ++t)
#pragma unroll
    for (int r = 0; r < 8; ++r) {
      int qrow = q0 + w * 16 + r + 8 * khalf;
      int col  = t * 16 + l16;
      Y[((long)(b * T_ + qrow)) * C_ + h * HD_ + col] = (bf16)(o[t][r] * invl[r]);
    }
}

// ---------------------------------------------------------------- launch
extern "C" void kernel_launch(void* const* d_in, const int* in_sizes, int n_in,
                              void* d_out, int out_size, void* d_ws, size_t ws_size,
                              hipStream_t stream) {
  const float* x     = (const float*)d_in[0];
  const float* Wq    = (const float*)d_in[1];
  const float* Wk    = (const float*)d_in[2];
  const float* Wv    = (const float*)d_in[3];
  const float* Wproj = (const float*)d_in[4];

  char* ws = (char*)d_ws;
  size_t off = 0;
  auto alloc = [&](size_t bytes) -> void* {
    void* p = ws + off;
    off += (bytes + 255) & ~(size_t)255;
    return p;
  };
  bf16* xb  = (bf16*)alloc((size_t)M_ * C_ * 2);
  bf16* WqT = (bf16*)alloc((size_t)C_ * C_ * 2);
  bf16* WkT = (bf16*)alloc((size_t)C_ * NKVD * 2);
  bf16* WvT = (bf16*)alloc((size_t)C_ * NKVD * 2);
  bf16* WpT = (bf16*)alloc((size_t)C_ * C_ * 2);
  bf16* Qb  = (bf16*)alloc((size_t)M_ * C_ * 2);
  bf16* Kb  = (bf16*)alloc((size_t)M_ * NKVD * 2);
  bf16* Vb  = (bf16*)alloc((size_t)M_ * NKVD * 2);
  bf16* Yb  = (bf16*)alloc((size_t)M_ * C_ * 2);

  int nx = M_ * C_;
  f32_to_bf16_kernel<<<(nx + 255) / 256, 256, 0, stream>>>(x, xb, nx);
  transpose_to_bf16_kernel<<<(C_ * C_ + 255) / 256, 256, 0, stream>>>(Wq, WqT, C_, C_);
  transpose_to_bf16_kernel<<<(C_ * NKVD + 255) / 256, 256, 0, stream>>>(Wk, WkT, C_, NKVD);
  transpose_to_bf16_kernel<<<(C_ * NKVD + 255) / 256, 256, 0, stream>>>(Wv, WvT, C_, NKVD);
  transpose_to_bf16_kernel<<<(C_ * C_ + 255) / 256, 256, 0, stream>>>(Wproj, WpT, C_, C_);

  dim3 gq(C_ / BN, M_ / BM);
  gemm_bf16_kernel<<<gq, 256, 0, stream>>>(xb, WqT, Qb, M_, C_, C_, 0);
  dim3 gk(NKVD / BN, M_ / BM);
  gemm_bf16_kernel<<<gk, 256, 0, stream>>>(xb, WkT, Kb, M_, NKVD, C_, 0);
  gemm_bf16_kernel<<<gk, 256, 0, stream>>>(xb, WvT, Vb, M_, NKVD, C_, 0);

  float qscale = 0.08838834764831845f; // 1/sqrt(128)
  int nrq = M_ * NH_ * 64;
  rope_kernel<<<(nrq + 255) / 256, 256, 0, stream>>>(Qb, NH_, qscale);
  int nrk = M_ * NKV_ * 64;
  rope_kernel<<<(nrk + 255) / 256, 256, 0, stream>>>(Kb, NKV_, 1.0f);

  dim3 gf(T_ / BQ, NH_, B_);
  flash_attn_kernel<<<gf, 256, 0, stream>>>(Qb, Kb, Vb, Yb);

  gemm_bf16_kernel<<<gq, 256, 0, stream>>>(Yb, WpT, d_out, M_, C_, C_, 1);
}
